// AMRSRN_47485158425466
// MI455X (gfx1250) — matrix operand order repaired and checked by reading, not compiled, not run on hardware
//
#include <hip/hip_runtime.h>
#include <hip/hip_bf16.h>

#define G_   64
#define C_   2
#define DD   64
#define HH   64
#define WW   64
#define NPTS 262144
#define FIN  128   // G_*C_
#define RS   136   // LDS row stride in bf16 elems (128 + 16B pad -> kills bank conflicts)

typedef __attribute__((ext_vector_type(16))) __bf16 v16bf;
typedef __attribute__((ext_vector_type(8)))  __bf16 v8bf;
typedef __attribute__((ext_vector_type(8)))  float  v8f;

__device__ __forceinline__ unsigned short f2bf(float f) {
  return __builtin_bit_cast(unsigned short, (__bf16)f);   // v_cvt, RTNE
}
__device__ __forceinline__ float bf2f(unsigned short h) {
  unsigned u = ((unsigned)h) << 16;
  return __builtin_bit_cast(float, u);
}

// ---------------------------------------------------------------------------
// One-shot: convert all four 128x128 f32 weight matrices to bf16 in workspace
// so 2048 MLP blocks stage pure bf16 (2x less global traffic, no VALU conv).
// ---------------------------------------------------------------------------
__global__ __launch_bounds__(256) void amr_convw(
    const float* __restrict__ W0, const float* __restrict__ W1,
    const float* __restrict__ W2, const float* __restrict__ W3,
    unsigned short* __restrict__ wbf)
{
  const int i = blockIdx.x * 256 + threadIdx.x;   // 0 .. 65535
  const int l = i >> 14;
  const int j = i & 16383;
  const float* src = (l == 0) ? W0 : (l == 1) ? W1 : (l == 2) ? W2 : W3;
  wbf[i] = f2bf(src[j]);
}

// ---------------------------------------------------------------------------
// Phase 1: trilinear sampling of 64 grids x 2 channels -> bf16 feature matrix
// thread = (point, grid); 64 consecutive lanes share a point -> packed u32
// stores (2 bf16 channels) are fully coalesced. Grids (134MB) live in L2.
// ---------------------------------------------------------------------------
__global__ __launch_bounds__(256) void amr_sample(
    const float* __restrict__ x,
    const float* __restrict__ scales,
    const float* __restrict__ trans,
    const float* __restrict__ grids,
    unsigned int* __restrict__ feats32)
{
  const int t = blockIdx.x * 256 + threadIdx.x;
  const int g = t & (G_ - 1);
  const int p = t >> 6;

  const float px = x[p*3+0], py = x[p*3+1], pz = x[p*3+2];
  const float sx = scales[g*3+0], sy = scales[g*3+1], sz = scales[g*3+2];
  const float ox = trans[g*3+0],  oy = trans[g*3+1],  oz = trans[g*3+2];

  const float fx = (px*sx + ox + 1.f) * (0.5f * (WW - 1));
  const float fy = (py*sy + oy + 1.f) * (0.5f * (HH - 1));
  const float fz = (pz*sz + oz + 1.f) * (0.5f * (DD - 1));
  const float xf = floorf(fx), yf = floorf(fy), zf = floorf(fz);
  const float tx = fx - xf, ty = fy - yf, tz = fz - zf;
  const int x0 = (int)xf, y0 = (int)yf, z0 = (int)zf;

  const float* __restrict__ vol = grids + (size_t)g * (C_*DD*HH*WW);
  const int CH = DD*HH*WW;

  const float wxv[2] = {1.f - tx, tx};
  const float wyv[2] = {1.f - ty, ty};
  const float wzv[2] = {1.f - tz, tz};
  const int xs[2] = {x0, x0+1}, ys[2] = {y0, y0+1}, zs[2] = {z0, z0+1};

  float a0 = 0.f, a1 = 0.f;
  #pragma unroll
  for (int iz = 0; iz < 2; ++iz)
  #pragma unroll
  for (int iy = 0; iy < 2; ++iy)
  #pragma unroll
  for (int ix = 0; ix < 2; ++ix) {
    const int X = xs[ix], Y = ys[iy], Z = zs[iz];
    const bool ok = (X >= 0) & (X < WW) & (Y >= 0) & (Y < HH) & (Z >= 0) & (Z < DD);
    const int Xc = min(max(X, 0), WW-1);
    const int Yc = min(max(Y, 0), HH-1);
    const int Zc = min(max(Z, 0), DD-1);
    float w = wzv[iz] * wyv[iy] * wxv[ix];
    w = ok ? w : 0.f;
    const size_t idx = ((size_t)Zc * HH + Yc) * WW + Xc;
    a0 += vol[idx]      * w;
    a1 += vol[CH + idx] * w;
  }
  const unsigned pack = (unsigned)f2bf(a0) | ((unsigned)f2bf(a1) << 16);
  feats32[(size_t)p * G_ + g] = pack;   // columns 2g, 2g+1 of the 128-wide row
}

// ---------------------------------------------------------------------------
// Phase 2: fused 4-layer MLP + final projection, bf16 WMMA, f32 accum.
// Block = 128 points x 8 waves; wave w owns activation rows [16w,16w+16):
// exactly its A-operand rows each layer -> wave-local LDS handoff.
// K-outer / N-inner with 8 resident accumulators: 8 independent WMMA chains
// back-to-back per K step, A fragment reused 8x.
// ---------------------------------------------------------------------------
__global__ __launch_bounds__(256) void amr_mlp(
    const unsigned short* __restrict__ feats,
    const unsigned short* __restrict__ wbf,   // 4 x 128x128 bf16, (N,K) row-major
    const float* __restrict__ b0, const float* __restrict__ b1,
    const float* __restrict__ b2, const float* __restrict__ b3,
    const float* __restrict__ Wf, const float* __restrict__ bfin,
    float* __restrict__ out)
{
  __shared__ __align__(16) unsigned short wLDS[128 * RS];
  __shared__ __align__(16) unsigned short actLDS[128 * RS];
  __shared__ float wfLDS[128];

  const int tid  = threadIdx.x;
  const int lane = tid & 31;
  const int wave = tid >> 5;
  const int hi   = lane >> 4;      // half-wave selector per ISA fragment layout
  const int l16  = lane & 15;
  const int mbase = wave * 16;
  const int blockRow = blockIdx.x * 128;

  // stage this block's 128x128 bf16 feature tile into LDS (16B vector copies)
  for (int i = tid; i < 128 * 16; i += 256) {
    const int r = i >> 4, c = i & 15;
    const uint4 v = ((const uint4*)(feats + (size_t)(blockRow + r) * FIN))[c];
    *((uint4*)&actLDS[r * RS + c * 8]) = v;
  }
  if (tid < 128) wfLDS[tid] = Wf[tid];

  auto stageW = [&](int layer) {
    const unsigned short* src = wbf + layer * (128 * 128);
    for (int i = tid; i < 128 * 16; i += 256) {
      const int r = i >> 4, c = i & 15;
      const uint4 v = ((const uint4*)(src + r * 128))[c];
      *((uint4*)&wLDS[r * RS + c * 8]) = v;
    }
  };

  // A fragment: 16x32 bf16, lane m=l16; hi=0 -> K{0..7,16..23}, hi=1 -> K{8..15,24..31}
  auto loadA = [&](int m, int kt) -> v16bf {
    const unsigned short* p = &actLDS[m * RS + kt * 32];
    v8bf lo = *(const v8bf*)(p + 8 * hi);
    v8bf hv = *(const v8bf*)(p + 16 + 8 * hi);
    return __builtin_shufflevector(lo, hv, 0,1,2,3,4,5,6,7,8,9,10,11,12,13,14,15);
  };
  // B fragment: 32x16, lane n=l16; hi=0 -> K{0..7,16..23}, hi=1 -> K{8..15,24..31} in W row n
  auto loadB = [&](int nt, int kt) -> v16bf {
    const unsigned short* p = &wLDS[(nt * 16 + l16) * RS + kt * 32];
    v8bf lo = *(const v8bf*)(p + 8 * hi);
    v8bf hv = *(const v8bf*)(p + 16 + 8 * hi);
    return __builtin_shufflevector(lo, hv, 0,1,2,3,4,5,6,7,8,9,10,11,12,13,14,15);
  };

  auto run_layer = [&](const float* __restrict__ bias, int act_kind) {
    v8f acc[8];
    #pragma unroll
    for (int nt = 0; nt < 8; ++nt) {
      const float bv = bias[nt * 16 + l16];        // C init = bias broadcast over M
      acc[nt] = (v8f){bv, bv, bv, bv, bv, bv, bv, bv};
    }
    #pragma unroll
    for (int k = 0; k < 4; ++k) {
      const v16bf A = loadA(mbase + l16, k);
      #pragma unroll
      for (int nt = 0; nt < 8; ++nt) {             // 8 independent WMMA chains
        v16bf B = loadB(nt, k);
        acc[nt] = __builtin_amdgcn_wmma_f32_16x16x32_bf16(
                      false, A, false, B, (short)0, acc[nt], false, false);
      }
    }
    // D layout: vgpr r -> M = r + 8*hi ; N = l16. Activation + bf16 writeback.
    // All A reads finished (k loop done) before actLDS is overwritten in place.
    #pragma unroll
    for (int nt = 0; nt < 8; ++nt) {
      #pragma unroll
      for (int r = 0; r < 8; ++r) {
        float y = acc[nt][r];
        if (act_kind == 0) { float s = __sinf(y); y = 0.5f * y + s * s; }
        else               { y = (y >= 0.f) ? y : 0.2f * y; }
        actLDS[(mbase + r + 8 * hi) * RS + nt * 16 + l16] = f2bf(y);
      }
    }
  };

  stageW(0);
  __builtin_prefetch(wbf + 1 * 128 * 128 + tid, 0, 0);  // global_prefetch_b8
  __syncthreads();
  run_layer(b0, 0);
  __syncthreads();
  stageW(1);
  __builtin_prefetch(wbf + 2 * 128 * 128 + tid, 0, 0);
  __syncthreads();
  run_layer(b1, 1);
  __syncthreads();
  stageW(2);
  __builtin_prefetch(wbf + 3 * 128 * 128 + tid, 0, 0);
  __syncthreads();
  run_layer(b2, 1);
  __syncthreads();
  stageW(3);
  __syncthreads();
  run_layer(b3, 1);

  // final 128->1 projection: 2 lanes per row; rows stay wave-local (t/2 in
  // [16w,16w+16) for lanes of wave w) so in-order DS makes stores visible.
  const int r  = tid >> 1;
  const int kh = tid & 1;
  const unsigned short* arow = &actLDS[r * RS + kh * 64];
  const float* wf = &wfLDS[kh * 64];
  float sum = 0.f;
  #pragma unroll
  for (int k = 0; k < 64; ++k) sum += bf2f(arow[k]) * wf[k];
  sum += __shfl_xor(sum, 1, 32);
  if (kh == 0) out[blockRow + r] = sum + bfin[0];
}

// ---------------------------------------------------------------------------
extern "C" void kernel_launch(void* const* d_in, const int* in_sizes, int n_in,
                              void* d_out, int out_size, void* d_ws, size_t ws_size,
                              hipStream_t stream) {
  const float* x      = (const float*)d_in[0];
  const float* scales = (const float*)d_in[1];
  const float* trans  = (const float*)d_in[2];
  const float* grids  = (const float*)d_in[3];
  const float* W0 = (const float*)d_in[4];  const float* b0 = (const float*)d_in[5];
  const float* W1 = (const float*)d_in[6];  const float* b1 = (const float*)d_in[7];
  const float* W2 = (const float*)d_in[8];  const float* b2 = (const float*)d_in[9];
  const float* W3 = (const float*)d_in[10]; const float* b3 = (const float*)d_in[11];
  const float* Wf = (const float*)d_in[12]; const float* bf = (const float*)d_in[13];

  unsigned char* ws = (unsigned char*)d_ws;
  unsigned int*   feats32 = (unsigned int*)ws;                      // 64 MB
  unsigned short* feats16 = (unsigned short*)ws;
  unsigned short* wbf     = (unsigned short*)(ws + (size_t)NPTS * FIN * 2);  // 128 KB

  amr_convw<<<(4 * 128 * 128) / 256, 256, 0, stream>>>(W0, W1, W2, W3, wbf);
  amr_sample<<<(NPTS * G_) / 256, 256, 0, stream>>>(x, scales, trans, grids, feats32);
  amr_mlp<<<NPTS / 128, 256, 0, stream>>>(feats16, wbf, b0, b1, b2, b3,
                                          Wf, bf, (float*)d_out);
}
